// MoEFFN_86723979641511
// MI455X (gfx1250) — matrix operand order repaired and checked by reading, not compiled, not run on hardware
//
#include <hip/hip_runtime.h>
#include <stdint.h>

// MoE FFN: B=4, T=2048, D=1024, F=2752, E=8, TOPK=2
#define N_TOK 8192
#define DIM   1024
#define FF    2752
#define NEXP  8
#define NSLOT 16384                 // N_TOK * TOPK
#define BM    64                    // M rows per block (64-aligned expert ranges)
#define KU    64                    // K elements staged per LDS round (2 WMMA K-steps)
#define NSEG  264                   // ceil((NSLOT + NEXP*BM)/BM)
#define NPAD  (NSEG * BM)           // 16896 padded gathered rows
#define LSTR  72                    // LDS row stride (ushorts): 144B, 16B-aligned, bank-staggered

typedef __attribute__((ext_vector_type(8)))  float  f32x8;
typedef __attribute__((ext_vector_type(8)))  __bf16 bf16x8;
typedef __attribute__((ext_vector_type(16))) __bf16 bf16x16;

// ---------------- workspace layout (bytes) ----------------
static constexpr size_t OFF_TOPI   = 0;                         // int[NSLOT]
static constexpr size_t OFF_TOPW   = 65536;                     // float[NSLOT]
static constexpr size_t OFF_META   = 131072;                    // int[64]: [0..7]=counts [8..16]=offs [17..24]=cursors
static constexpr size_t OFF_SEG    = 131328;                    // int[NSEG]
static constexpr size_t OFF_ROWMAP = 132608;                    // int[NPAD]
static constexpr size_t OFF_XB     = 200192;                    // bf16[N_TOK*DIM]   (16.8 MB)
static constexpr size_t OFF_HB     = 16977408;                  // bf16[NPAD*FF]     (93.0 MB)

__device__ __forceinline__ unsigned short f32_to_bf16(float f) {
    unsigned int u = __float_as_uint(f);
    u += 0x7FFFu + ((u >> 16) & 1u);          // round-to-nearest-even
    return (unsigned short)(u >> 16);
}

__device__ __forceinline__ bf16x16 frag16(const unsigned short* p0, const unsigned short* p1) {
    bf16x8 lo = *(const bf16x8*)p0;           // 16-byte LDS load
    bf16x8 hi = *(const bf16x8*)p1;           // 16-byte LDS load
    return __builtin_shufflevector(lo, hi, 0,1,2,3,4,5,6,7,8,9,10,11,12,13,14,15);
}

// A 16x32 bf16 fragment at K offset j0: chunks at j0+grp*8 and j0+16+grp*8
__device__ __forceinline__ bf16x16 fragA(const unsigned short* rowbase, int j0, int grp) {
    return frag16(rowbase + j0 + grp * 8, rowbase + j0 + 16 + grp * 8);
}
// B 32x16 bf16 fragment at K offset j0: chunks at j0+grp*8 and j0+16+grp*8
__device__ __forceinline__ bf16x16 fragB(const unsigned short* colbase, int j0, int grp) {
    const unsigned short* p = colbase + j0 + grp * 8;
    return frag16(p, p + 16);
}

// ---------------- 1) router: logits, softmax top-2, counts ----------------
__global__ void moe_router(const float* __restrict__ x, const float* __restrict__ Wr,
                           int* __restrict__ topi, float* __restrict__ topw,
                           int* __restrict__ meta) {
    const int wave = threadIdx.x >> 5;
    const int lane = threadIdx.x & 31;
    const int n    = blockIdx.x * 8 + wave;   // one wave32 per token
    float acc[NEXP];
#pragma unroll
    for (int e = 0; e < NEXP; ++e) acc[e] = 0.f;
    const float* xr = x + (size_t)n * DIM;
    for (int i = 0; i < DIM / 32; ++i) {
        const int d = i * 32 + lane;
        const float xv = xr[d];
        const float4* wr = (const float4*)(Wr + (size_t)d * NEXP);
        const float4 w0 = wr[0], w1 = wr[1];
        acc[0] += xv * w0.x; acc[1] += xv * w0.y; acc[2] += xv * w0.z; acc[3] += xv * w0.w;
        acc[4] += xv * w1.x; acc[5] += xv * w1.y; acc[6] += xv * w1.z; acc[7] += xv * w1.w;
    }
#pragma unroll
    for (int e = 0; e < NEXP; ++e) {
        float v = acc[e];
        for (int off = 16; off > 0; off >>= 1) v += __shfl_xor(v, off, 32);
        acc[e] = v;
    }
    if (lane == 0) {
        int i0 = 0; float m0 = acc[0];
#pragma unroll
        for (int e = 1; e < NEXP; ++e) if (acc[e] > m0) { m0 = acc[e]; i0 = e; }
        int i1 = -1; float m1 = -3.4e38f;
#pragma unroll
        for (int e = 0; e < NEXP; ++e) if (e != i0 && acc[e] > m1) { m1 = acc[e]; i1 = e; }
        const float p1 = __expf(m1 - m0);             // top-2 renormalized softmax
        const float w0 = 1.f / (1.f + p1);
        topi[2 * n]     = i0;  topi[2 * n + 1] = i1;
        topw[2 * n]     = w0;  topw[2 * n + 1] = 1.f - w0;
        atomicAdd(&meta[i0], 1);
        atomicAdd(&meta[i1], 1);
    }
}

// ---------------- 2) scan: 64-aligned offsets + segment->expert ----------------
__global__ void moe_scan(int* __restrict__ meta, int* __restrict__ seg) {
    int total = 0;
    for (int e = 0; e < NEXP; ++e) {
        meta[8 + e] = total;
        total += ((meta[e] + BM - 1) / BM) * BM;
    }
    meta[8 + NEXP] = total;
    for (int b = 0; b < NSEG; ++b) seg[b] = -1;
    for (int e = 0; e < NEXP; ++e)
        for (int b = meta[8 + e] / BM; b < meta[8 + e + 1] / BM; ++b) seg[b] = e;
}

__global__ void moe_fill(int* __restrict__ rowmap) {
    const int i = blockIdx.x * 256 + threadIdx.x;
    if (i < NPAD) rowmap[i] = -1;
}

__global__ void moe_place(const int* __restrict__ topi, int* __restrict__ meta,
                          int* __restrict__ rowmap) {
    const int i = blockIdx.x * 256 + threadIdx.x;       // slot = n*2+k
    const int e = topi[i];
    const int pos = atomicAdd(&meta[17 + e], 1);        // order-independent values
    rowmap[meta[8 + e] + pos] = i;
}

// ---------------- 3) x -> bf16 ----------------
__global__ void moe_cvt_x(const float* __restrict__ x, unsigned short* __restrict__ xb) {
    const int i = (blockIdx.x * 256 + threadIdx.x) * 4;
    const float4 v = *(const float4*)(x + i);
    *(ushort4*)(xb + i) = make_ushort4(f32_to_bf16(v.x), f32_to_bf16(v.y),
                                       f32_to_bf16(v.z), f32_to_bf16(v.w));
}

// ---------------- 4) gathered GEMM1: h = silu(x W1) * (x W3) ----------------
// Block tile: 64 rows x 32 h-cols (gate+up), K staged 64 at a time.
__global__ void __launch_bounds__(256) moe_gemm13(
    const float* __restrict__ W13, const unsigned short* __restrict__ xb,
    const int* __restrict__ rowmap, const int* __restrict__ seg,
    unsigned short* __restrict__ hb) {
    const int e = seg[blockIdx.x];
    if (e < 0) return;
    __shared__ unsigned short aL[BM * LSTR];            // 64 rows x 64 K
    __shared__ unsigned short bgL[32 * LSTR];           // 32 cols x 64 K (col-major)
    __shared__ unsigned short buL[32 * LSTR];
    __shared__ int rmL[BM];
    const int tid  = threadIdx.x;
    const int row0 = blockIdx.x * BM;
    const int f0   = blockIdx.y * 32;
    if (tid < BM) rmL[tid] = rowmap[row0 + tid];
    __syncthreads();
    const float* We = W13 + (size_t)e * DIM * (2 * FF);
    const int wave = tid >> 5, lane = tid & 31;
    const int rt = wave >> 1,  ct = wave & 1;
    const int grp = lane >> 4, lr = lane & 15;
    const int ar = tid >> 2,   ac = (tid & 3) * 16;     // A staging: 16 bf16 per thread
    const int br = tid >> 2,   bc = (tid & 3) * 8;      // B staging: 8 fp32 per matrix
    const int tokA = rmL[ar];
    f32x8 accg = {0.f,0.f,0.f,0.f,0.f,0.f,0.f,0.f};
    f32x8 accu = accg;
    for (int kk = 0; kk < DIM; kk += KU) {
        uint4 av0 = make_uint4(0u, 0u, 0u, 0u);         // pad rows -> zeros -> h=0
        uint4 av1 = av0;
        if (tokA >= 0) {
            const unsigned short* xp = xb + (size_t)(tokA >> 1) * DIM + kk + ac;
            av0 = *(const uint4*)xp;
            av1 = *(const uint4*)(xp + 8);
        }
        *(uint4*)(&aL[ar * LSTR + ac])     = av0;
        *(uint4*)(&aL[ar * LSTR + ac + 8]) = av1;
        const float* pg = We + (size_t)(kk + br) * (2 * FF) + f0 + bc;
        const float4 g4a = *(const float4*)pg;
        const float4 g4b = *(const float4*)(pg + 4);
        const float4 u4a = *(const float4*)(pg + FF);
        const float4 u4b = *(const float4*)(pg + FF + 4);
        if (kk + KU < DIM) __builtin_prefetch(pg + (size_t)KU * 2 * FF, 0, 1);
        bgL[(bc + 0) * LSTR + br] = f32_to_bf16(g4a.x);
        bgL[(bc + 1) * LSTR + br] = f32_to_bf16(g4a.y);
        bgL[(bc + 2) * LSTR + br] = f32_to_bf16(g4a.z);
        bgL[(bc + 3) * LSTR + br] = f32_to_bf16(g4a.w);
        bgL[(bc + 4) * LSTR + br] = f32_to_bf16(g4b.x);
        bgL[(bc + 5) * LSTR + br] = f32_to_bf16(g4b.y);
        bgL[(bc + 6) * LSTR + br] = f32_to_bf16(g4b.z);
        bgL[(bc + 7) * LSTR + br] = f32_to_bf16(g4b.w);
        buL[(bc + 0) * LSTR + br] = f32_to_bf16(u4a.x);
        buL[(bc + 1) * LSTR + br] = f32_to_bf16(u4a.y);
        buL[(bc + 2) * LSTR + br] = f32_to_bf16(u4a.z);
        buL[(bc + 3) * LSTR + br] = f32_to_bf16(u4a.w);
        buL[(bc + 4) * LSTR + br] = f32_to_bf16(u4b.x);
        buL[(bc + 5) * LSTR + br] = f32_to_bf16(u4b.y);
        buL[(bc + 6) * LSTR + br] = f32_to_bf16(u4b.z);
        buL[(bc + 7) * LSTR + br] = f32_to_bf16(u4b.w);
        __syncthreads();
        const unsigned short* ab = &aL[(rt * 16 + lr) * LSTR];
        const unsigned short* gb = &bgL[(ct * 16 + lr) * LSTR];
        const unsigned short* ub = &buL[(ct * 16 + lr) * LSTR];
#pragma unroll
        for (int j0 = 0; j0 < KU; j0 += 32) {           // 2 WMMA K-steps per staged tile
            const bf16x16 af = fragA(ab, j0, grp);
            const bf16x16 gf = fragB(gb, j0, grp);
            const bf16x16 uf = fragB(ub, j0, grp);
            accg = __builtin_amdgcn_wmma_f32_16x16x32_bf16(false, af, false, gf, (short)0, accg, false, false);
            accu = __builtin_amdgcn_wmma_f32_16x16x32_bf16(false, af, false, uf, (short)0, accu, false, false);
        }
        __syncthreads();
    }
    const int col = f0 + ct * 16 + lr;
#pragma unroll
    for (int j = 0; j < 8; ++j) {                       // C layout: M = j + 8*grp, N = lr
        const int srow = row0 + rt * 16 + grp * 8 + j;
        const float g = accg[j], u = accu[j];
        const float h = g * u * __builtin_amdgcn_rcpf(1.f + __expf(-g));  // SwiGLU via v_rcp_f32
        hb[(size_t)srow * FF + col] = f32_to_bf16(h);
    }
}

// ---------------- 5) gathered GEMM2: out += w * (h W2), 2 commutative atomics ----------------
// Block tile: 64 rows x 64 out-cols; each wave owns 2 col tiles (A-frag reuse).
__global__ void __launch_bounds__(256) moe_gemm2(
    const float* __restrict__ W2, const unsigned short* __restrict__ hb,
    const int* __restrict__ rowmap, const int* __restrict__ seg,
    const float* __restrict__ topw, float* __restrict__ out) {
    const int e = seg[blockIdx.x];
    if (e < 0) return;
    __shared__ unsigned short aL[BM * LSTR];            // 64 rows x 64 K
    __shared__ unsigned short bL[64 * LSTR];            // 64 cols x 64 K (col-major)
    __shared__ int rmL[BM];
    const int tid  = threadIdx.x;
    const int row0 = blockIdx.x * BM;
    const int c0   = blockIdx.y * 64;
    if (tid < BM) rmL[tid] = rowmap[row0 + tid];
    __syncthreads();
    const float* We = W2 + (size_t)e * FF * DIM;
    const int wave = tid >> 5, lane = tid & 31;
    const int rt = wave >> 1,  cp = wave & 1;           // col pair: tiles cp*2, cp*2+1
    const int grp = lane >> 4, lr = lane & 15;
    const int ar = tid >> 2,   ac = (tid & 3) * 16;     // A staging: 16 bf16 per thread
    const int br = tid >> 2,   bc = (tid & 3) * 16;     // B staging: 16 fp32 per thread
    f32x8 acc0 = {0.f,0.f,0.f,0.f,0.f,0.f,0.f,0.f};
    f32x8 acc1 = acc0;
    for (int kk = 0; kk < FF; kk += KU) {
        const unsigned short* hp = hb + (size_t)(row0 + ar) * FF + kk + ac;
        const uint4 av0 = *(const uint4*)hp;
        const uint4 av1 = *(const uint4*)(hp + 8);
        *(uint4*)(&aL[ar * LSTR + ac])     = av0;
        *(uint4*)(&aL[ar * LSTR + ac + 8]) = av1;
        const float* pb = We + (size_t)(kk + br) * DIM + c0 + bc;
        if (kk + KU < FF) __builtin_prefetch(pb + (size_t)KU * DIM, 0, 1);
#pragma unroll
        for (int q = 0; q < 4; ++q) {
            const float4 b4 = *(const float4*)(pb + q * 4);
            bL[(bc + q * 4 + 0) * LSTR + br] = f32_to_bf16(b4.x);
            bL[(bc + q * 4 + 1) * LSTR + br] = f32_to_bf16(b4.y);
            bL[(bc + q * 4 + 2) * LSTR + br] = f32_to_bf16(b4.z);
            bL[(bc + q * 4 + 3) * LSTR + br] = f32_to_bf16(b4.w);
        }
        __syncthreads();
        const unsigned short* ab = &aL[(rt * 16 + lr) * LSTR];
        const unsigned short* b0 = &bL[(cp * 32 + lr) * LSTR];
        const unsigned short* b1 = &bL[(cp * 32 + 16 + lr) * LSTR];
#pragma unroll
        for (int j0 = 0; j0 < KU; j0 += 32) {           // 2 WMMA K-steps per staged tile
            const bf16x16 af  = fragA(ab, j0, grp);
            const bf16x16 bf0 = fragB(b0, j0, grp);
            const bf16x16 bf1 = fragB(b1, j0, grp);
            acc0 = __builtin_amdgcn_wmma_f32_16x16x32_bf16(false, af, false, bf0, (short)0, acc0, false, false);
            acc1 = __builtin_amdgcn_wmma_f32_16x16x32_bf16(false, af, false, bf1, (short)0, acc1, false, false);
        }
        __syncthreads();
    }
    const int col0 = c0 + cp * 32 + lr;
#pragma unroll
    for (int j = 0; j < 8; ++j) {
        const int m  = rt * 16 + grp * 8 + j;
        const int rm = rmL[m];
        if (rm >= 0) {
            const float w = topw[rm];
            float* orow = out + (size_t)(rm >> 1) * DIM;
            atomicAdd(&orow[col0],      w * acc0[j]);
            atomicAdd(&orow[col0 + 16], w * acc1[j]);
        }
    }
}

extern "C" void kernel_launch(void* const* d_in, const int* in_sizes, int n_in,
                              void* d_out, int out_size, void* d_ws, size_t ws_size,
                              hipStream_t stream) {
    (void)in_sizes; (void)n_in; (void)ws_size;
    const float* x   = (const float*)d_in[0];
    const float* W13 = (const float*)d_in[1];
    const float* W2  = (const float*)d_in[2];
    const float* Wr  = (const float*)d_in[3];
    float* out = (float*)d_out;
    char*  ws  = (char*)d_ws;

    int*   topi   = (int*)(ws + OFF_TOPI);
    float* topw   = (float*)(ws + OFF_TOPW);
    int*   meta   = (int*)(ws + OFF_META);
    int*   seg    = (int*)(ws + OFF_SEG);
    int*   rowmap = (int*)(ws + OFF_ROWMAP);
    unsigned short* xb = (unsigned short*)(ws + OFF_XB);
    unsigned short* hb = (unsigned short*)(ws + OFF_HB);

    hipMemsetAsync(meta, 0, 256, stream);
    hipMemsetAsync(out, 0, (size_t)out_size * sizeof(float), stream);

    moe_router<<<N_TOK / 8, 256, 0, stream>>>(x, Wr, topi, topw, meta);
    moe_scan  <<<1, 1, 0, stream>>>(meta, seg);
    moe_fill  <<<(NPAD + 255) / 256, 256, 0, stream>>>(rowmap);
    moe_place <<<NSLOT / 256, 256, 0, stream>>>(topi, meta, rowmap);
    moe_cvt_x <<<(N_TOK * DIM / 4) / 256, 256, 0, stream>>>(x, xb);

    dim3 g1(NSEG, FF / 32);     // 264 x 86
    moe_gemm13<<<g1, 256, 0, stream>>>(W13, xb, rowmap, seg, hb);
    dim3 g2(NSEG, DIM / 64);    // 264 x 16
    moe_gemm2 <<<g2, 256, 0, stream>>>(W2, hb, rowmap, seg, topw, out);
}
